// EdgeAttention_49289044689430
// MI455X (gfx1250) — compile-verified, matrix operand order checked by reading, not run on hardware
//
#include <hip/hip_runtime.h>
#include <hip/hip_bf16.h>
#include <math.h>

#define BB 32
#define LL 256
#define DD 256
#define EE 32
#define HH 8
#define HD 32
#define SCALE_F 0.17677669529663687f  // 1/sqrt(32)

typedef __attribute__((ext_vector_type(16))) __bf16 v16bf;
typedef __attribute__((ext_vector_type(8)))  __bf16 v8bf;
typedef __attribute__((ext_vector_type(8)))  float  v8f;
typedef __attribute__((ext_vector_type(4)))  float  v4f;
typedef __attribute__((ext_vector_type(2)))  float  v2f;

// Build a 16x32 bf16 WMMA A/B fragment from a row-major f32 source.
// Lane (half*16 + r) holds row r, K = {half*8+0..7, 16+half*8+0..7}.
static __device__ __forceinline__ v16bf frag_from_f32(const float* base, int stride) {
  const int lane = threadIdx.x & 31;
  const int r = lane & 15;
  const int g = (lane >> 4) & 1;
  const float* row = base + r * stride + g * 8;
  v4f a0 = *(const v4f*)(row);
  v4f a1 = *(const v4f*)(row + 4);
  v4f a2 = *(const v4f*)(row + 16);
  v4f a3 = *(const v4f*)(row + 20);
  v16bf out;
#pragma unroll
  for (int i = 0; i < 4; ++i) {
    out[i]      = (__bf16)a0[i];
    out[4 + i]  = (__bf16)a1[i];
    out[8 + i]  = (__bf16)a2[i];
    out[12 + i] = (__bf16)a3[i];
  }
  return out;
}

// Same fragment from a row-major bf16 source: two 16B loads per lane.
static __device__ __forceinline__ v16bf frag_from_bf16(const __bf16* base, int stride) {
  const int lane = threadIdx.x & 31;
  const int r = lane & 15;
  const int g = (lane >> 4) & 1;
  const __bf16* row = base + r * stride;
  v8bf lo = *(const v8bf*)(row + g * 8);
  v8bf hi = *(const v8bf*)(row + 16 + g * 8);
  v16bf out;
#pragma unroll
  for (int i = 0; i < 8; ++i) { out[i] = lo[i]; out[8 + i] = hi[i]; }
  return out;
}

// ---------------------------------------------------------------------------
// Kernel 1: packed QKV projection. qkv = x @ W^T + b, routed to
//   q (pre-scaled by 1/sqrt(HD)) -> [B,H,L,HD] bf16
//   k                            -> [B,H,L,HD] bf16
//   v (transposed)               -> [B,H,HD,L] bf16
// One wave per 16x16 output tile; K=256 in 8 steps of bf16 WMMA.
// ---------------------------------------------------------------------------
__global__ __launch_bounds__(256) void qkv_proj_kernel(
    const float* __restrict__ x, const float* __restrict__ w,
    const float* __restrict__ bias,
    __bf16* __restrict__ qbf, __bf16* __restrict__ kbf, __bf16* __restrict__ vt) {
  const int wave = blockIdx.x * 8 + (threadIdx.x >> 5);
  const int NT = (3 * DD) / 16;  // 48
  const int mt = wave / NT, nt = wave % NT;
  const int lane = threadIdx.x & 31, r = lane & 15, g = lane >> 4;

  v8f acc;
  const float bv = bias[nt * 16 + r];
#pragma unroll
  for (int i = 0; i < 8; ++i) acc[i] = bv;

#pragma unroll 1
  for (int kc = 0; kc < DD / 32; ++kc) {
    v16bf a  = frag_from_f32(x + (mt * 16) * DD + kc * 32, DD);
    v16bf bf = frag_from_f32(w + (nt * 16) * DD + kc * 32, DD);
    acc = __builtin_amdgcn_wmma_f32_16x16x32_bf16(false, a, false, bf, (short)0, acc, false, false);
  }

#pragma unroll
  for (int rr = 0; rr < 8; ++rr) {
    const int m = mt * 16 + rr + g * 8;    // row in [B*L]
    const int n = nt * 16 + r;             // col in [3*D]
    const int b = m >> 8, l = m & 255;
    const int sec = n >> 8, c = n & 255, h = c >> 5, d = c & 31;
    const float v = acc[rr];
    if (sec == 0)      qbf[((b * HH + h) * LL + l) * HD + d] = (__bf16)(v * SCALE_F);
    else if (sec == 1) kbf[((b * HH + h) * LL + l) * HD + d] = (__bf16)v;
    else               vt [((b * HH + h) * HD + d) * LL + l] = (__bf16)v;
  }
}

// ---------------------------------------------------------------------------
// Kernel 2: fused edge attention for one (batch, 16-row q tile).
//  Phase A: one bf16 WMMA per 16-pair tile computes e (cols 0-7) and the
//           gate pre-activation (cols 8-15) from edge_x; e -> S (LDS f32),
//           sigmoid(gate) -> G (LDS bf16).
//  Phase B: wave h: S[h] = q_h @ k_h^T (q pre-scaled) + e  (bf16 WMMA, C=e).
//  Phase C: edge_out = S_pairs @ eo_w^T + eo_b via f32 WMMA 16x16x4 (K=H=8).
//  Phase D: wave h: row softmax over 256 keys, * gate, attn @ v (bf16 WMMA),
//           result bf16 -> abf [B,L,D].
// ---------------------------------------------------------------------------
__global__ __launch_bounds__(256) void edge_attn_kernel(
    const float* __restrict__ edge_x,
    const __bf16* __restrict__ qbf, const __bf16* __restrict__ kbf,
    const __bf16* __restrict__ vt,
    const float* __restrict__ e_w, const float* __restrict__ e_b,
    const float* __restrict__ g_w, const float* __restrict__ g_b,
    const float* __restrict__ eo_w, const float* __restrict__ eo_b,
    float* __restrict__ edge_out, __bf16* __restrict__ abf) {
  __shared__ __align__(16) float  S[HH][16][LL];  // 128 KB: e -> scores -> weights
  __shared__ __align__(16) __bf16 G[HH][16][LL];  //  64 KB: sigmoid gate

  const int b    = blockIdx.x >> 4;
  const int q0   = (blockIdx.x & 15) * 16;
  const int wave = threadIdx.x >> 5;
  const int lane = threadIdx.x & 31, r = lane & 15, g = lane >> 4;

  // ---- Phase A: e + gate from edge_x -------------------------------------
  v16bf wfrag;  // B operand: cols 0-7 = e_w rows, cols 8-15 = g_w rows (K=E=32)
  {
    const float* wrow = (r < 8) ? (e_w + r * EE) : (g_w + (r - 8) * EE);
#pragma unroll
    for (int i = 0; i < 8; ++i) {
      wfrag[i]     = (__bf16)wrow[g * 8 + i];
      wfrag[8 + i] = (__bf16)wrow[16 + g * 8 + i];
    }
  }
  const float biasn = (r < 8) ? e_b[r] : g_b[r - 8];

#pragma unroll 1
  for (int t = wave; t < 256; t += 8) {       // 16 q-rows x 16 k-tiles
    const int q = t >> 4, k0 = (t & 15) * 16;
    v16bf a = frag_from_f32(edge_x + ((b * LL + (q0 + q)) * LL + k0) * EE, EE);
    v8f acc;
#pragma unroll
    for (int i = 0; i < 8; ++i) acc[i] = biasn;
    acc = __builtin_amdgcn_wmma_f32_16x16x32_bf16(false, a, false, wfrag, (short)0, acc, false, false);
#pragma unroll
    for (int rr = 0; rr < 8; ++rr) {
      const int kc = k0 + rr + g * 8;         // D row = k column within tile
      if (r < 8) S[r][q][kc] = acc[rr];
      else       G[r - 8][q][kc] = (__bf16)(1.0f / (1.0f + __expf(-acc[rr])));
    }
  }
  __syncthreads();

  // ---- Phase B: raw scores (wave == head) --------------------------------
  {
    const int h = wave;
    v16bf qf = frag_from_bf16(qbf + ((b * HH + h) * LL + q0) * HD, HD);
#pragma unroll 1
    for (int kt = 0; kt < 16; ++kt) {
      v16bf kf = frag_from_bf16(kbf + ((b * HH + h) * LL + kt * 16) * HD, HD);
      v8f c;
#pragma unroll
      for (int rr = 0; rr < 8; ++rr) c[rr] = S[h][rr + g * 8][kt * 16 + r];
      c = __builtin_amdgcn_wmma_f32_16x16x32_bf16(false, qf, false, kf, (short)0, c, false, false);
#pragma unroll
      for (int rr = 0; rr < 8; ++rr) S[h][rr + g * 8][kt * 16 + r] = c[rr];
    }
  }
  __syncthreads();

  // ---- Phase C: edge_out from raw scores (f32 WMMA, K=H=8) ---------------
  v2f bfr[2][2];      // [K-step][N-tile]; B[k][n] = eo_w[n][k]
  float ebias[2];
#pragma unroll
  for (int nt = 0; nt < 2; ++nt) {
    ebias[nt] = eo_b[nt * 16 + r];
#pragma unroll
    for (int st = 0; st < 2; ++st) {
      v2f tmp;
#pragma unroll
      for (int j = 0; j < 2; ++j) tmp[j] = eo_w[(nt * 16 + r) * HH + st * 4 + g * 2 + j];
      bfr[st][nt] = tmp;
    }
  }
#pragma unroll 1
  for (int t = wave; t < 256; t += 8) {
    const int q = t >> 4, k0 = (t & 15) * 16;
    v2f af[2];        // A[m][k]: m = k column, k = head
#pragma unroll
    for (int st = 0; st < 2; ++st) {
      v2f tmp;
#pragma unroll
      for (int j = 0; j < 2; ++j) tmp[j] = S[st * 4 + g * 2 + j][q][k0 + r];
      af[st] = tmp;
    }
#pragma unroll
    for (int nt = 0; nt < 2; ++nt) {
      v8f acc;
#pragma unroll
      for (int i = 0; i < 8; ++i) acc[i] = ebias[nt];
      acc = __builtin_amdgcn_wmma_f32_16x16x4_f32(false, af[0], false, bfr[0][nt], (short)0, acc, false, false);
      acc = __builtin_amdgcn_wmma_f32_16x16x4_f32(false, af[1], false, bfr[1][nt], (short)0, acc, false, false);
#pragma unroll
      for (int rr = 0; rr < 8; ++rr) {
        const int kc = k0 + rr + g * 8;
        edge_out[((b * LL + q0 + q) * LL + kc) * EE + nt * 16 + r] = acc[rr];
      }
    }
  }
  __syncthreads();

  // ---- Phase D: softmax * gate, attn @ v (wave == head) ------------------
  {
    const int h = wave;
#pragma unroll 1
    for (int q = 0; q < 16; ++q) {
      float vals[8];
      float mx = -__builtin_inff();
#pragma unroll
      for (int j = 0; j < 8; ++j) { vals[j] = S[h][q][lane + j * 32]; mx = fmaxf(mx, vals[j]); }
#pragma unroll
      for (int off = 16; off >= 1; off >>= 1) mx = fmaxf(mx, __shfl_xor(mx, off, 32));
      float sum = 0.f;
#pragma unroll
      for (int j = 0; j < 8; ++j) { vals[j] = __expf(vals[j] - mx); sum += vals[j]; }
#pragma unroll
      for (int off = 16; off >= 1; off >>= 1) sum += __shfl_xor(sum, off, 32);
      const float inv = 1.0f / sum;
#pragma unroll
      for (int j = 0; j < 8; ++j) {
        const int k = lane + j * 32;
        S[h][q][k] = vals[j] * inv * (float)G[h][q][k];
      }
    }
#pragma unroll
    for (int nt = 0; nt < 2; ++nt) {
      v8f acc;
#pragma unroll
      for (int i = 0; i < 8; ++i) acc[i] = 0.0f;
#pragma unroll 1
      for (int kc = 0; kc < 8; ++kc) {
        v16bf a  = frag_from_f32(&S[h][0][kc * 32], LL);
        v16bf vf = frag_from_bf16(vt + ((b * HH + h) * HD + nt * 16) * LL + kc * 32, LL);
        acc = __builtin_amdgcn_wmma_f32_16x16x32_bf16(false, a, false, vf, (short)0, acc, false, false);
      }
#pragma unroll
      for (int rr = 0; rr < 8; ++rr) {
        const int l = q0 + rr + g * 8;
        abf[(b * LL + l) * DD + h * HD + nt * 16 + r] = (__bf16)acc[rr];
      }
    }
  }
}

// ---------------------------------------------------------------------------
// Kernel 3: output projection out = a @ out_w^T + out_b  (f32 result)
// ---------------------------------------------------------------------------
__global__ __launch_bounds__(256) void out_proj_kernel(
    const __bf16* __restrict__ abf, const float* __restrict__ w,
    const float* __restrict__ bias, float* __restrict__ out) {
  const int wave = blockIdx.x * 8 + (threadIdx.x >> 5);
  const int NT = DD / 16;  // 16
  const int mt = wave / NT, nt = wave % NT;
  const int lane = threadIdx.x & 31, r = lane & 15, g = lane >> 4;

  v8f acc;
  const float bv = bias[nt * 16 + r];
#pragma unroll
  for (int i = 0; i < 8; ++i) acc[i] = bv;

#pragma unroll 1
  for (int kc = 0; kc < DD / 32; ++kc) {
    v16bf a  = frag_from_bf16(abf + (mt * 16) * DD + kc * 32, DD);
    v16bf bf = frag_from_f32(w + (nt * 16) * DD + kc * 32, DD);
    acc = __builtin_amdgcn_wmma_f32_16x16x32_bf16(false, a, false, bf, (short)0, acc, false, false);
  }
#pragma unroll
  for (int rr = 0; rr < 8; ++rr)
    out[(mt * 16 + rr + g * 8) * DD + nt * 16 + r] = acc[rr];
}

extern "C" void kernel_launch(void* const* d_in, const int* in_sizes, int n_in,
                              void* d_out, int out_size, void* d_ws, size_t ws_size,
                              hipStream_t stream) {
  const float* x      = (const float*)d_in[0];
  const float* edge_x = (const float*)d_in[1];
  const float* in_w   = (const float*)d_in[2];
  const float* in_b   = (const float*)d_in[3];
  const float* out_w  = (const float*)d_in[4];
  const float* out_b  = (const float*)d_in[5];
  const float* e_w    = (const float*)d_in[6];
  const float* e_b    = (const float*)d_in[7];
  const float* g_w    = (const float*)d_in[8];
  const float* g_b    = (const float*)d_in[9];
  const float* eo_w   = (const float*)d_in[10];
  const float* eo_b   = (const float*)d_in[11];

  float* out_main = (float*)d_out;                          // [B,L,D]
  float* edge_out = out_main + (size_t)BB * LL * DD;        // [B,L,L,E]

  char* ws = (char*)d_ws;                                   // 16 MB used
  __bf16* qbf = (__bf16*)(ws);                              // [B,H,L,HD] 4MB
  __bf16* kbf = (__bf16*)(ws + ((size_t)4 << 20));          // [B,H,L,HD] 4MB
  __bf16* vt  = (__bf16*)(ws + ((size_t)8 << 20));          // [B,H,HD,L] 4MB
  __bf16* abf = (__bf16*)(ws + ((size_t)12 << 20));         // [B,L,D]    4MB

  // (B*L/16) * (3D/16) tiles / 8 waves = 3072 blocks
  qkv_proj_kernel<<<3072, 256, 0, stream>>>(x, in_w, in_b, qbf, kbf, vt);
  // one block per (batch, 16-row q tile)
  edge_attn_kernel<<<BB * (LL / 16), 256, 0, stream>>>(
      edge_x, qbf, kbf, vt, e_w, e_b, g_w, g_b, eo_w, eo_b, edge_out, abf);
  // (B*L/16) * (D/16) tiles / 8 waves = 1024 blocks
  out_proj_kernel<<<1024, 256, 0, stream>>>(abf, out_w, out_b, out_main);
}
